// Decoder_69337952026902
// MI455X (gfx1250) — matrix operand order repaired and checked by reading, not compiled, not run on hardware
//
#include <hip/hip_runtime.h>
#include <cmath>

typedef __attribute__((ext_vector_type(16))) _Float16 v16h;
typedef __attribute__((ext_vector_type(8)))  _Float16 v8h;
typedef __attribute__((ext_vector_type(8)))  float    v8f;
typedef __attribute__((ext_vector_type(4)))  float    v4f;

// ---------------- problem dims ----------------
#define BB   512
#define LL   256
#define II   64
#define H0A  269        // N_INTER actual
#define H0P  288        // padded (multiple of 32)
#define H1A  179        // N_COMMAND actual
#define H1P  192
#define H2A  64         // N_MOTOR
#define H2P  64
// cell0: K = 64 + 288 = 352 -> pad 384, N = 4*288 = 1152
#define KT0  12
#define NT0  72
#define NPW0 9
// cell1: K = 288 + 192 = 480 -> pad 512, N = 768
#define KT1  16
#define NT1  48
#define NPW1 6
// cell2: K = 192 + 64 = 256, N = 256
#define KT2  8
#define NT2  16
#define NPW2 2
// LDS X widths/strides (strides chosen for bank spread; stride*2/4 mod 64 == 4)
#define XW0  384
#define XS0  392
#define XW1  512
#define XS1  520
#define XS2  264

static __device__ inline v8f zero8() {
    v8f r;
#pragma unroll
    for (int i = 0; i < 8; ++i) r[i] = 0.0f;
    return r;
}

// Compose a 16x32 f16 A-fragment from LDS, ISA 7.12.2 layout.
static __device__ inline v16h make_a(const _Float16* p) {
    v8h lo = *(const v8h*)p;
    v8h hi = *(const v8h*)(p + 16);
    v16h a;
#pragma unroll
    for (int i = 0; i < 8; ++i) { a[i] = lo[i]; a[i + 8] = hi[i]; }
    return a;
}

// Load one half-strip of packed B fragments (2*HK x b128, immediate offsets).
template <int HK>
static __device__ inline void load_half(v16h (&b)[HK], const _Float16* __restrict__ wp) {
#pragma unroll
    for (int k = 0; k < HK; ++k) b[k] = *(const v16h*)(wp + k * 512);
}

// WMMA over one half-strip; A fragments pipelined depth-2 from LDS.
template <int HK, int KOFF>
static __device__ inline v8f half_wmma(const _Float16* __restrict__ xrow,
                                       const v16h (&b)[HK], v8f acc) {
    v16h a0 = make_a(xrow + KOFF * 32);
    v16h a1 = (HK > 1) ? make_a(xrow + (KOFF + 1) * 32) : a0;
#pragma unroll
    for (int k = 0; k < HK; ++k) {
        acc = __builtin_amdgcn_wmma_f32_16x16x32_f16(false, a0, false, b[k],
                                                     (short)0, acc, false, false);
        a0 = a1;
        if (k + 2 < HK) a1 = make_a(xrow + (KOFF + k + 2) * 32);
    }
    return acc;
}

// Tile-contiguous Z store: lane writes its 8 accumulators as 2 x b128.
static __device__ inline void store_tile(float* __restrict__ Zt, int nt, int lane, v8f acc) {
    float* p = Zt + nt * 256 + lane * 8;
    v4f lo, hi;
#pragma unroll
    for (int i = 0; i < 4; ++i) { lo[i] = acc[i]; hi[i] = acc[i + 4]; }
    *(v4f*)p       = lo;
    *(v4f*)(p + 4) = hi;
}

// Index into tile-contiguous Z for logical element (row m, padded col ng).
static __device__ inline int zidx(int m, int ng) {
    int nt   = ng >> 4;
    int lane = ((m & 8) ? 16 : 0) + (ng & 15);
    return nt * 256 + lane * 8 + (m & 7);
}

// One cell GEMM, half-strip ping-pong pipeline, pressure <= ~200 VGPRs.
// sched_barrier(0) pins the 4-stage shape so the scheduler cannot inflate it.
template <int KT, int NPW>
static __device__ inline void cell_gemm(const _Float16* __restrict__ X, int xs,
                                        const _Float16* __restrict__ Wp,
                                        float* __restrict__ Zt,
                                        int wave, int lane) {
    constexpr int HK = KT / 2;
    const int m    = lane & 15;
    const int koff = (lane & 16) ? 8 : 0;
    const int ntbase = wave * NPW;

    const _Float16* __restrict__ xrow = X + m * xs + koff;
    const _Float16* __restrict__ wpL  = Wp + (size_t)ntbase * KT * 512 + lane * 16;

    v16h bP[HK], bQ[HK];
    load_half<HK>(bP, wpL);                               // tile 0, half 0
    for (int i = 0; i < NPW; ++i) {                       // rolled
        const _Float16* wq = wpL + ((size_t)i * KT + HK) * 512;
        load_half<HK>(bQ, wq);                            // tile i, half 1 in flight
        __builtin_amdgcn_sched_barrier(0);
        v8f acc = half_wmma<HK, 0>(xrow, bP, zero8());    // consume half 0
        __builtin_amdgcn_sched_barrier(0);
        load_half<HK>(bP, wq + (size_t)HK * 512);         // tile i+1, half 0 (last overreads 1 half: safe)
        __builtin_amdgcn_sched_barrier(0);
        acc = half_wmma<HK, HK>(xrow, bQ, acc);           // consume half 1
        store_tile(Zt, ntbase + i, lane, acc);
        __builtin_amdgcn_sched_barrier(0);
    }
}

static __device__ inline float ncp_act(float ff1, float ff2, float ta, float tb) {
    float ti = 1.0f / (1.0f + expf(-(ta + tb)));
    return tanhf(ff1) * (1.0f - ti) + ti * tanhf(ff2);
}

// ------------- weight packing (masked, padded, B-fragment order) -------------
__global__ void pack_w_kernel(const float* __restrict__ W, const float* __restrict__ M,
                              _Float16* __restrict__ Wp,
                              int NT, int KT, int h, int hp, int Wld,
                              int LA, int PA, int LB) {
    int idx = blockIdx.x * blockDim.x + threadIdx.x;
    int total = NT * KT * 512;
    if (idx >= total) return;
    int e    = idx & 511;
    int kt   = (idx >> 9) % KT;
    int nt   = idx / (512 * KT);
    int lane = e >> 4;
    int j    = e & 15;
    int n_p  = nt * 16 + (lane & 15);
    int k_p  = kt * 32 + ((lane & 16) ? 8 : 0) + (j & 7) + ((j & 8) ? 16 : 0);
    float v = 0.0f;
    int q = n_p / hp, r = n_p % hp;
    int c = -1;
    if (k_p < LA) c = k_p;
    else if (k_p >= PA && k_p < PA + LB) c = LA + (k_p - PA);
    if (r < h && c >= 0) {
        int row = q * h + r;
        v = W[(size_t)row * Wld + c];
        if (M != nullptr && q < 2) v *= M[(size_t)r * Wld + c];
    }
    Wp[idx] = (_Float16)v;
}

__global__ void pack_b_kernel(const float* __restrict__ b, float* __restrict__ bp,
                              int h, int hp) {
    int idx = blockIdx.x * blockDim.x + threadIdx.x;
    if (idx >= 4 * hp) return;
    int q = idx / hp, n = idx % hp;
    bp[idx] = (n < h) ? b[q * h + n] : 0.0f;
}

// ------------------------------ main RNN kernel ------------------------------
__global__ void __launch_bounds__(256)
ncp_rnn_kernel(const float* __restrict__ dt, const float* __restrict__ hidden,
               const _Float16* __restrict__ Wp0, const _Float16* __restrict__ Wp1,
               const _Float16* __restrict__ Wp2, const _Float16* __restrict__ WpO,
               const float* __restrict__ bp0, const float* __restrict__ bp1,
               const float* __restrict__ bp2, const float* __restrict__ bout,
               float* __restrict__ out) {
    __shared__ __align__(16) _Float16 sX0[16 * XS0]; // [dt(64) | h0(288) | 0-pad(32)]
    __shared__ __align__(16) _Float16 sX1[16 * XS1]; // [a(288)  | h1(192) | 0-pad(32)]
    __shared__ __align__(16) _Float16 sX2[16 * XS2]; // [b(192)  | h2(64)]
    __shared__ __align__(16) float    sZ[NT0 * 256]; // tile-contiguous Z (max 72 tiles)

    const int tid  = threadIdx.x;
    const int lane = tid & 31;
    const int wave = tid >> 5;
    const int b0r  = blockIdx.x * 16;

    // init hidden state (f32 -> f16, zero padding incl. K-pad columns)
    for (int idx = tid; idx < 16 * (XW0 - II); idx += 256) {
        int m = idx / (XW0 - II), n = idx % (XW0 - II);
        float v = (n < H0A) ? hidden[(size_t)(b0r + m) * 512 + n] : 0.0f;
        sX0[m * XS0 + II + n] = (_Float16)v;
    }
    for (int idx = tid; idx < 16 * (XW1 - H0P); idx += 256) {
        int m = idx / (XW1 - H0P), n = idx % (XW1 - H0P);
        float v = (n < H1A) ? hidden[(size_t)(b0r + m) * 512 + H0A + n] : 0.0f;
        sX1[m * XS1 + H0P + n] = (_Float16)v;
    }
    for (int idx = tid; idx < 16 * H2P; idx += 256) {
        int m = idx / H2P, n = idx % H2P;
        float v = hidden[(size_t)(b0r + m) * 512 + H0A + H1A + n];
        sX2[m * XS2 + H1P + n] = (_Float16)v;
    }
    __syncthreads();

    for (int t = 0; t < LL; ++t) {
        // stage dt_t
        for (int idx = tid; idx < 16 * II; idx += 256) {
            int m = idx >> 6, i = idx & 63;
            sX0[m * XS0 + i] = (_Float16)dt[((size_t)(b0r + m) * LL + t) * II + i];
        }
        __syncthreads();

        // ---- cell 0 ----
        cell_gemm<KT0, NPW0>(sX0, XS0, Wp0, sZ, wave, lane);
        __syncthreads();
        for (int idx = tid; idx < 16 * H0P; idx += 256) {
            int m = idx / H0P, n = idx % H0P;
            float o = 0.0f;
            if (n < H0A) {
                o = ncp_act(sZ[zidx(m, n)] + bp0[n],
                            sZ[zidx(m, H0P + n)] + bp0[H0P + n],
                            sZ[zidx(m, 2 * H0P + n)] + bp0[2 * H0P + n],
                            sZ[zidx(m, 3 * H0P + n)] + bp0[3 * H0P + n]);
            }
            _Float16 oh = (_Float16)o;
            sX1[m * XS1 + n] = oh;           // input of cell1
            sX0[m * XS0 + II + n] = oh;      // recurrent h0
        }
        __syncthreads();

        // ---- cell 1 ----
        cell_gemm<KT1, NPW1>(sX1, XS1, Wp1, sZ, wave, lane);
        __syncthreads();
        for (int idx = tid; idx < 16 * H1P; idx += 256) {
            int m = idx / H1P, n = idx % H1P;
            float o = 0.0f;
            if (n < H1A) {
                o = ncp_act(sZ[zidx(m, n)] + bp1[n],
                            sZ[zidx(m, H1P + n)] + bp1[H1P + n],
                            sZ[zidx(m, 2 * H1P + n)] + bp1[2 * H1P + n],
                            sZ[zidx(m, 3 * H1P + n)] + bp1[3 * H1P + n]);
            }
            _Float16 oh = (_Float16)o;
            sX2[m * XS2 + n] = oh;           // input of cell2
            sX1[m * XS1 + H0P + n] = oh;     // recurrent h1
        }
        __syncthreads();

        // ---- cell 2 ----
        cell_gemm<KT2, NPW2>(sX2, XS2, Wp2, sZ, wave, lane);
        __syncthreads();
        for (int idx = tid; idx < 16 * H2P; idx += 256) {
            int m = idx / H2P, n = idx % H2P;
            float o = ncp_act(sZ[zidx(m, n)] + bp2[n],
                              sZ[zidx(m, H2P + n)] + bp2[H2P + n],
                              sZ[zidx(m, 2 * H2P + n)] + bp2[2 * H2P + n],
                              sZ[zidx(m, 3 * H2P + n)] + bp2[3 * H2P + n]);
            sX2[m * XS2 + H1P + n] = (_Float16)o;   // recurrent h2 == c_t
        }
        __syncthreads();

        // ---- output projection: y_t = c_t @ Wout^T + bout (waves 0..3) ----
        if (wave < 4) {
            const int m    = lane & 15;
            const int koff = (lane & 16) ? 8 : 0;
            v16h a0 = make_a(sX2 + m * XS2 + H1P + koff);
            v16h a1 = make_a(sX2 + m * XS2 + H1P + 32 + koff);
            const _Float16* __restrict__ wp = WpO + (size_t)wave * 2 * 512 + lane * 16;
            v16h bb0 = *(const v16h*)(wp);
            v16h bb1 = *(const v16h*)(wp + 512);
            v8f acc = zero8();
            acc = __builtin_amdgcn_wmma_f32_16x16x32_f16(false, a0, false, bb0,
                                                         (short)0, acc, false, false);
            acc = __builtin_amdgcn_wmma_f32_16x16x32_f16(false, a1, false, bb1,
                                                         (short)0, acc, false, false);
            const int mrow = (lane & 16) ? 8 : 0;
            const int col  = wave * 16 + m;
            const float bv = bout[col];
#pragma unroll
            for (int v = 0; v < 8; ++v)
                out[((size_t)(b0r + mrow + v) * LL + t) * 64 + col] = acc[v] + bv;
        }
        __syncthreads();
    }
}

// --------------------------------- launcher ---------------------------------
static inline int cdiv(int a, int b) { return (a + b - 1) / b; }

extern "C" void kernel_launch(void* const* d_in, const int* in_sizes, int n_in,
                              void* d_out, int out_size, void* d_ws, size_t ws_size,
                              hipStream_t stream) {
    const float* dt     = (const float*)d_in[0];
    const float* hidden = (const float*)d_in[1];
    const float* W0     = (const float*)d_in[2];
    const float* b0     = (const float*)d_in[3];
    const float* W1     = (const float*)d_in[4];
    const float* b1     = (const float*)d_in[5];
    const float* W2     = (const float*)d_in[6];
    const float* b2     = (const float*)d_in[7];
    const float* Wout   = (const float*)d_in[8];
    const float* bout   = (const float*)d_in[9];
    const float* M0     = (const float*)d_in[10];
    const float* M1     = (const float*)d_in[11];
    const float* M2     = (const float*)d_in[12];
    float* out = (float*)d_out;

    char* ws = (char*)d_ws;
    size_t o = 0;
    _Float16* Wp0 = (_Float16*)(ws + o); o += (size_t)NT0 * KT0 * 512 * 2;
    _Float16* Wp1 = (_Float16*)(ws + o); o += (size_t)NT1 * KT1 * 512 * 2;
    _Float16* Wp2 = (_Float16*)(ws + o); o += (size_t)NT2 * KT2 * 512 * 2;
    _Float16* WpO = (_Float16*)(ws + o); o += (size_t)4 * 2 * 512 * 2;
    float* bp0 = (float*)(ws + o); o += 4 * H0P * 4;
    float* bp1 = (float*)(ws + o); o += 4 * H1P * 4;
    float* bp2 = (float*)(ws + o); o += 4 * H2P * 4;
    o += (size_t)KT1 * 512 * 2;  // slack for harmless half-strip overread

    // pack weights into WMMA B-fragment order (masked + padded; K-pad -> zeros)
    pack_w_kernel<<<cdiv(NT0 * KT0 * 512, 256), 256, 0, stream>>>(
        W0, M0, Wp0, NT0, KT0, H0A, H0P, II + H0A, II, II, H0A);
    pack_w_kernel<<<cdiv(NT1 * KT1 * 512, 256), 256, 0, stream>>>(
        W1, M1, Wp1, NT1, KT1, H1A, H1P, H0A + H1A, H0A, H0P, H1A);
    pack_w_kernel<<<cdiv(NT2 * KT2 * 512, 256), 256, 0, stream>>>(
        W2, M2, Wp2, NT2, KT2, H2A, H2P, H1A + H2A, H1A, H1P, H2A);
    pack_w_kernel<<<cdiv(4 * 2 * 512, 256), 256, 0, stream>>>(
        Wout, nullptr, WpO, 4, 2, 64, 64, 64, 64, 64, 0);

    pack_b_kernel<<<cdiv(4 * H0P, 256), 256, 0, stream>>>(b0, bp0, H0A, H0P);
    pack_b_kernel<<<cdiv(4 * H1P, 256), 256, 0, stream>>>(b1, bp1, H1A, H1P);
    pack_b_kernel<<<cdiv(4 * H2P, 256), 256, 0, stream>>>(b2, bp2, H2A, H2P);

    // persistent RNN: one block per 16-row batch tile
    ncp_rnn_kernel<<<BB / 16, 256, 0, stream>>>(
        dt, hidden, Wp0, Wp1, Wp2, WpO, bp0, bp1, bp2, bout, out);
}
